// Keyattention_56487409877252
// MI455X (gfx1250) — matrix operand order repaired and checked by reading, not compiled, not run on hardware
//
#include <hip/hip_runtime.h>

// Problem constants (from reference): B=16, C=6, P=10, FD=256, D=128, H=W=128
#define NB    16
#define NC    6
#define NP    10
#define FD    256
#define DD    128
#define HWTOT 16384
#define TILE  128
#define CP    60     // C*P
#define CPPAD 64     // padded to multiple of 16 for WMMA

#define XSTRIDE 264  // bf16 elements per XbfT row (128 pixels x 264): pad vs bank conflicts
#define KSTRIDE 136  // bf16 elements per keyT row (128 pixels x 136)

typedef __attribute__((ext_vector_type(16))) __bf16 v16bf;
typedef __attribute__((ext_vector_type(8)))  __bf16 v8bf;
typedef __attribute__((ext_vector_type(2)))  __bf16 v2bf;
typedef __attribute__((ext_vector_type(8)))  float  v8f;
typedef __attribute__((ext_vector_type(4)))  float  v4f;

// ---------------------------------------------------------------------------
// Kernel A: q[b,cp,d] = Wq[c] @ prototypes[b,c,p] + bq[c]
//   - fp32 result straight into the protos_out section of d_out (exact)
//   - bf16 copy, zero-padded to 64 rows, into workspace for the WMMA kernel
// grid: NB*CPPAD blocks x 128 threads (one thread per d)
// ---------------------------------------------------------------------------
__global__ void qproj_kernel(const float* __restrict__ protos,
                             const float* __restrict__ Wq,
                             const float* __restrict__ bq,
                             float* __restrict__ protosOut,
                             __bf16* __restrict__ qbf) {
  int b  = blockIdx.x / CPPAD;
  int cp = blockIdx.x % CPPAD;
  int d  = threadIdx.x;
  float v = 0.0f;
  if (cp < CP) {
    int c = cp / NP;
    const float* w = Wq     + (size_t)(c * DD + d) * FD;
    const float* x = protos + (size_t)(b * CP + cp) * FD;   // [B,C,P,FD] row = b*60+cp
    float acc = bq[c * DD + d];
#pragma unroll 8
    for (int f = 0; f < FD; ++f) acc += w[f] * x[f];
    v = acc;
    protosOut[(size_t)(b * CP + cp) * DD + d] = acc;
  }
  qbf[(size_t)(b * CPPAD + cp) * DD + d] = (__bf16)v;
}

// ---------------------------------------------------------------------------
// Kernel B: fused key-GEMM -> sim-GEMM -> softmax-max -> feature rescale.
// One block = one (batch, 128-pixel tile). 256 threads = 8 waves.
// Dynamic LDS (~101 KB): XbfT bf16 [128][264] | keyT bf16 [128][136] | wmax[128]
// Feature tile is staged once as transposed bf16 (row pairs packed into one
// b32 store via v_cvt_pk_bf16_f32) so every WMMA B-fragment is two contiguous
// ds_load_b128s; the fp32 tile is re-read from cache for the final rescale.
// ---------------------------------------------------------------------------
__launch_bounds__(256)
__global__ void keyatt_kernel(const float* __restrict__ assp,
                              const float* __restrict__ Wk,
                              const float* __restrict__ bk,
                              const __bf16* __restrict__ qbf,
                              float* __restrict__ outW) {
  extern __shared__ __align__(16) char smem[];
  __bf16* XbfT = (__bf16*)smem;                    // [TILE][XSTRIDE]  67584 B
  __bf16* keyT = XbfT + TILE * XSTRIDE;            // [TILE][KSTRIDE]  34816 B
  float*  wmax = (float*)(keyT + TILE * KSTRIDE);  // [TILE]             512 B

  const int b    = blockIdx.x >> 7;
  const int tile = blockIdx.x & 127;
  const int n0g  = tile << 7;
  const int tid  = threadIdx.x;
  const int lane = tid & 31;
  const int wv   = tid >> 5;             // wave id 0..7
  const bool lo  = (lane < 16);

  const float* src = assp + (size_t)b * FD * HWTOT + n0g;

  // Prefetch next tile of this batch toward L2 (gfx1250 global_prefetch_b8).
  if (tile < 127) {
    const float* nxt = src + TILE + (size_t)tid * HWTOT;
    __builtin_prefetch(nxt, 0, 3);
    __builtin_prefetch(nxt + 64, 0, 3);
  }

  // ---- Stage feature tile: global fp32 -> LDS bf16, transposed ----
  // Each thread handles 4 pixels x 2 consecutive rows per iteration so the
  // transposed write is one packed (r, r+1) b32 store per pixel.
  {
    int n  = (tid & 31) * 4;
    int r0 = (tid >> 5) * 2;
#pragma unroll
    for (int r = r0; r < FD; r += 16) {
      v4f xa = *(const v4f*)(src + (size_t)r * HWTOT + n);
      v4f xb = *(const v4f*)(src + (size_t)(r + 1) * HWTOT + n);
#pragma unroll
      for (int j = 0; j < 4; ++j) {
        v2bf pk;
        pk[0] = (__bf16)xa[j];
        pk[1] = (__bf16)xb[j];
        *(v2bf*)(&XbfT[(n + j) * XSTRIDE + r]) = pk;
      }
    }
  }
  __syncthreads();

  // ---- GEMM1: key[d, n] = Wk @ X + bk  (bf16 WMMA, f32 accumulate) ----
  {
    const int m0   = wv * 16;               // this wave's d-block
    const int arow = m0 + (lane & 15);      // A-matrix row held by this lane
    const int ak   = lo ? 0 : 16;           // A K sub-offset per lane half
    const float* wkrow = Wk + (size_t)arow * FD;

    v16bf afr[8];                           // all K-blocks of Wk, kept in VGPRs
#pragma unroll
    for (int kb = 0; kb < 8; ++kb) {
      const int k0 = kb * 32;
      v16bf a;
#pragma unroll
      for (int i = 0; i < 8; ++i) a[i]     = (__bf16)wkrow[k0 + ak + i];
#pragma unroll
      for (int i = 0; i < 8; ++i) a[8 + i] = (__bf16)wkrow[k0 + ak + 8 + i];
      afr[kb] = a;
    }

    const int hi8 = lo ? 0 : 8;
    float bias[8];
#pragma unroll
    for (int r = 0; r < 8; ++r) bias[r] = bk[m0 + hi8 + r];

    const int bk16 = lo ? 0 : 16;           // B K sub-offset per lane half
#pragma unroll 1
    for (int nb = 0; nb < 8; ++nb) {
      const int nn = nb * 16 + (lane & 15); // B/C column held by this lane
      v8f c;
#pragma unroll
      for (int r = 0; r < 8; ++r) c[r] = bias[r];
#pragma unroll
      for (int kb = 0; kb < 8; ++kb) {
        const v8bf* p = (const v8bf*)(&XbfT[nn * XSTRIDE + kb * 32 + bk16]);
        v8bf b0 = p[0], b1 = p[1];
        v16bf bf = __builtin_shufflevector(b0, b1, 0, 1, 2, 3, 4, 5, 6, 7,
                                           8, 9, 10, 11, 12, 13, 14, 15);
        c = __builtin_amdgcn_wmma_f32_16x16x32_bf16(false, afr[kb], false, bf,
                                                    (short)0, c, false, false);
      }
      // store this C column transposed as packed bf16: keyT[n][d..d+7]
      v8bf kv;
#pragma unroll
      for (int r = 0; r < 8; ++r) kv[r] = (__bf16)c[r];
      *(v8bf*)(&keyT[nn * KSTRIDE + m0 + hi8]) = kv;
    }
  }
  __syncthreads();

  // ---- GEMM2: sim[cp, n] = q @ key for this wave's 16-pixel block ----
  {
    const int nn   = wv * 16 + (lane & 15);
    const int bk16 = lo ? 0 : 16;

    v16bf bfr[4];                            // key B-fragments, K = 0..127
#pragma unroll
    for (int kb = 0; kb < 4; ++kb) {
      const v8bf* p = (const v8bf*)(&keyT[nn * KSTRIDE + kb * 32 + bk16]);
      v8bf b0 = p[0], b1 = p[1];
      bfr[kb] = __builtin_shufflevector(b0, b1, 0, 1, 2, 3, 4, 5, 6, 7,
                                        8, 9, 10, 11, 12, 13, 14, 15);
    }

    const __bf16* qb = qbf + (size_t)b * CPPAD * DD;
    const int qr = (lane & 15);
    const int ak = lo ? 0 : 16;

    v8f simc[4];
#pragma unroll 1
    for (int mb = 0; mb < 4; ++mb) {
      const __bf16* qrow = qb + (size_t)(mb * 16 + qr) * DD;
      v8f c = {};
#pragma unroll
      for (int kb = 0; kb < 4; ++kb) {
        v8bf a0 = *(const v8bf*)(qrow + kb * 32 + ak);
        v8bf a1 = *(const v8bf*)(qrow + kb * 32 + ak + 8);
        v16bf a = __builtin_shufflevector(a0, a1, 0, 1, 2, 3, 4, 5, 6, 7,
                                          8, 9, 10, 11, 12, 13, 14, 15);
        c = __builtin_amdgcn_wmma_f32_16x16x32_bf16(false, a, false, bfr[kb],
                                                    (short)0, c, false, false);
      }
      simc[mb] = c;
    }

    // ---- softmax over cp: weight_max[n] = 1 / sum_j exp(sim_j - max_j) ----
    // Lane half 'lo' holds rows 16mb+r, half 'hi' rows 16mb+8+r; padded rows
    // 60..63 live in (hi, mb==3, r>=4) and are excluded.
    float mx = -3.0e38f;
#pragma unroll
    for (int mb = 0; mb < 4; ++mb)
#pragma unroll
      for (int r = 0; r < 8; ++r) {
        bool excl = (!lo) && (mb == 3) && (r >= 4);
        mx = fmaxf(mx, excl ? -3.0e38f : simc[mb][r]);
      }
    float s = 0.0f;
#pragma unroll
    for (int mb = 0; mb < 4; ++mb)
#pragma unroll
      for (int r = 0; r < 8; ++r) {
        bool excl = (!lo) && (mb == 3) && (r >= 4);
        s += excl ? 0.0f : __expf(simc[mb][r] - mx);
      }
    float mx2 = __shfl_xor(mx, 16, 32);
    float s2  = __shfl_xor(s,  16, 32);
    float M   = fmaxf(mx, mx2);
    float S   = s * __expf(mx - M) + s2 * __expf(mx2 - M);
    if (lo) wmax[wv * 16 + lane] = 1.0f / S;
  }
  __syncthreads();

  // ---- Rescale: re-read fp32 tile (cache-hot) and stream out (NT) ----
  {
    float* dst = outW + (size_t)b * FD * HWTOT + n0g;
    int n  = (tid & 31) * 4;
    int r0 = tid >> 5;
    v4f wvec = *(const v4f*)(&wmax[n]);
#pragma unroll
    for (int r = r0; r < FD; r += 8) {
      v4f x = __builtin_nontemporal_load((const v4f*)(src + (size_t)r * HWTOT + n));
      x *= wvec;
      __builtin_nontemporal_store(x, (v4f*)(dst + (size_t)r * HWTOT + n));
    }
  }
}

// ---------------------------------------------------------------------------
extern "C" void kernel_launch(void* const* d_in, const int* in_sizes, int n_in,
                              void* d_out, int out_size, void* d_ws, size_t ws_size,
                              hipStream_t stream) {
  const float* assp   = (const float*)d_in[0];
  const float* protos = (const float*)d_in[1];
  const float* Wk     = (const float*)d_in[2];
  const float* bkv    = (const float*)d_in[3];
  const float* Wq     = (const float*)d_in[4];
  const float* bq     = (const float*)d_in[5];

  float* out       = (float*)d_out;
  float* protosOut = out + (size_t)NB * FD * HWTOT;      // protos_out section
  __bf16* qws      = (__bf16*)d_ws;                      // bf16 q, [B,64,128]

  qproj_kernel<<<NB * CPPAD, DD, 0, stream>>>(protos, Wq, bq, protosOut, qws);

  size_t smem = (size_t)TILE * XSTRIDE * 2 + (size_t)TILE * KSTRIDE * 2 +
                (size_t)TILE * sizeof(float);            // ~101 KB
  keyatt_kernel<<<NB * (HWTOT / TILE), 256, smem, stream>>>(
      assp, Wk, bkv, qws, out);
}
